// MultiHeadBatched_57543971832430
// MI455X (gfx1250) — matrix operand order repaired and checked by reading, not compiled, not run on hardware
//
#include <hip/hip_runtime.h>
#include <hip/hip_bf16.h>
#include <math.h>

// ---------------------------------------------------------------------------
// MHA forward for MI455X (gfx1250): bf16 WMMA + async global->LDS staging.
//   B=4, S=2048, E=1024, H=16, D=64
// ---------------------------------------------------------------------------

typedef __bf16 bf16;
typedef __attribute__((ext_vector_type(16))) __bf16 v16bf;
typedef __attribute__((ext_vector_type(4)))  __bf16 v4bf;
typedef __attribute__((ext_vector_type(8)))  float  v8f;

#define EMB   1024
#define NH    16
#define HD    64
#define BATCH 4
#define SEQ   2048
#define MTOT  (BATCH*SEQ)   // 8192 rows

union ABf { v16bf v; uint4 u[2]; };

__device__ __forceinline__ v8f wmma_bf16(const v16bf a, const v16bf b, const v8f c) {
  return __builtin_amdgcn_wmma_f32_16x16x32_bf16(false, a, false, b, (short)0, c, false, false);
}

// 16B async copy: global (64-bit VGPR pair) -> LDS (32-bit byte address VGPR).
// LDS address of a __shared__ object = low 32 bits of its generic pointer
// (flat-aperture truncation rule, ISA 10.2). Tracked on ASYNCcnt.
__device__ __forceinline__ void async_copy_b128(void* lds_ptr, const void* gptr) {
  unsigned lds_off = (unsigned)(size_t)lds_ptr;
  asm volatile("global_load_async_to_lds_b128 %0, %1, off"
               :: "v"(lds_off), "v"(gptr) : "memory");
}
__device__ __forceinline__ void wait_async_le4() {
  asm volatile("s_wait_asynccnt 0x4" ::: "memory");
}
__device__ __forceinline__ void wait_async_le0() {
  asm volatile("s_wait_asynccnt 0x0" ::: "memory");
}
__device__ __forceinline__ void wait_ds0() {
  asm volatile("s_wait_dscnt 0x0" ::: "memory");
}

// ---------------------------------------------------------------------------
// fp32 -> bf16 bulk convert (vectorized x4)
// ---------------------------------------------------------------------------
__global__ __launch_bounds__(256) void cvt_bf16_kernel(const float* __restrict__ x,
                                                       bf16* __restrict__ y, int n4) {
  int i = blockIdx.x * blockDim.x + threadIdx.x;
  if (i < n4) {
    float4 f = ((const float4*)x)[i];
    v4bf o;
    o.x = (bf16)f.x; o.y = (bf16)f.y; o.z = (bf16)f.z; o.w = (bf16)f.w;
    ((v4bf*)y)[i] = o;
  }
}

// ---------------------------------------------------------------------------
// per-batch "any valid kv position" flag
// ---------------------------------------------------------------------------
__global__ __launch_bounds__(256) void mask_any_kernel(const int* __restrict__ mask,
                                                       int* __restrict__ anyv) {
  __shared__ int red;
  int b = blockIdx.x;
  if (threadIdx.x == 0) red = 0;
  __syncthreads();
  int acc = 0;
  for (int i = threadIdx.x; i < SEQ; i += blockDim.x) acc |= mask[b * SEQ + i];
  if (acc) atomicOr(&red, 1);
  __syncthreads();
  if (threadIdx.x == 0) anyv[b] = red;
}

// ---------------------------------------------------------------------------
// GEMM: Y[m][n] = sum_k X[m][k] * W[n][k] + bias[n]   (X:[M,1024] W:[1024,1024])
// Block = 256 threads (8 waves as 4x2), tile 128(M) x 128(N), K-slices of 32
// double-buffered in LDS via async copies. Grid = (M/128, N/128).
//   mode 0: bf16 [B,H,S,D]   mode 1: bf16 [B,H,D,S]   mode 2: f32 [M,1024]
// ---------------------------------------------------------------------------
__global__ __launch_bounds__(256) void proj_gemm_kernel(const bf16* __restrict__ X,
                                                        const bf16* __restrict__ W,
                                                        const float* __restrict__ bias,
                                                        void* __restrict__ Y, int mode) {
  __shared__ bf16 xs[2][128 * 32];   // 8 KB per buffer
  __shared__ bf16 wsb[2][128 * 32];  // 8 KB per buffer

  const int tid  = threadIdx.x;
  const int lane = tid & 31;
  const int l15  = lane & 15;
  const int half = lane >> 4;
  const int wave_m = (tid >> 5) >> 1;   // 0..3
  const int wave_n = (tid >> 5) & 1;    // 0..1

  const int m0 = blockIdx.x * 128;
  const int n0 = blockIdx.y * 128;

  // 512 16B-chunks per 128x32 tile; each thread owns 2 (x2 tiles = 4 asyncs/step)
  auto issue = [&](int buf, int k0) {
#pragma unroll
    for (int c = tid; c < 512; c += 256) {
      const int row = c >> 2, cc = (c & 3) * 8;
      async_copy_b128(&xs[buf][row * 32 + cc],  X + (size_t)(m0 + row) * EMB + k0 + cc);
      async_copy_b128(&wsb[buf][row * 32 + cc], W + (size_t)(n0 + row) * EMB + k0 + cc);
    }
  };

  v8f acc[2][4] = {};
  issue(0, 0);

  const int nk = EMB / 32;
  for (int ks = 0; ks < nk; ++ks) {
    const int cur = ks & 1;
    if (ks + 1 < nk) { issue(cur ^ 1, (ks + 1) * 32); wait_async_le4(); }
    else             { wait_async_le0(); }
    __syncthreads();

    // A fragments (2 m-subtiles) from LDS
    ABf a[2];
#pragma unroll
    for (int sub = 0; sub < 2; ++sub) {
      const bf16* p = &xs[cur][(wave_m * 32 + sub * 16 + l15) * 32 + half * 8];
      a[sub].u[0] = *(const uint4*)(p);
      a[sub].u[1] = *(const uint4*)(p + 16);
    }
    // B fragments (4 n-subtiles) from LDS, each reused across both A frags
#pragma unroll
    for (int nf = 0; nf < 4; ++nf) {
      const bf16* p = &wsb[cur][(wave_n * 64 + nf * 16 + l15) * 32 + half * 8];
      ABf b;
      b.u[0] = *(const uint4*)(p);
      b.u[1] = *(const uint4*)(p + 16);
#pragma unroll
      for (int sub = 0; sub < 2; ++sub)
        acc[sub][nf] = wmma_bf16(a[sub].v, b.v, acc[sub][nf]);
    }
    __syncthreads();   // protect buffers before next overwrite
  }

#pragma unroll
  for (int sub = 0; sub < 2; ++sub)
#pragma unroll
    for (int nf = 0; nf < 4; ++nf) {
      const int n  = n0 + wave_n * 64 + nf * 16 + l15;
      const float bv = bias[n];
#pragma unroll
      for (int r = 0; r < 8; ++r) {
        const int m = m0 + wave_m * 32 + sub * 16 + r + half * 8;
        const float y = acc[sub][nf][r] + bv;
        if (mode == 2) {
          ((float*)Y)[(size_t)m * EMB + n] = y;
        } else {
          const int b = m >> 11, s = m & 2047;
          const int h = n >> 6,  d = n & 63;
          size_t off;
          if (mode == 0) off = (((size_t)(b * NH + h)) * SEQ + s) * HD + d;   // [B,H,S,D]
          else           off = (((size_t)(b * NH + h)) * HD + d) * SEQ + s;   // [B,H,D,S]
          ((bf16*)Y)[off] = (bf16)y;
        }
      }
    }
}

// ---------------------------------------------------------------------------
// Flash attention. Block = 4 waves sharing one (b,h); each wave owns 16 q rows
// x D=64. K/V tiles for each 32-wide kv step staged once per block into
// double-buffered LDS via async copies.
//   Qh,Kh: [B,H,S,D]   Vt: [B,H,D,S]   Oh: [B,S,H,D]
// ---------------------------------------------------------------------------
__global__ __launch_bounds__(128) void attn_kernel(const bf16* __restrict__ Qh,
                                                   const bf16* __restrict__ Kh,
                                                   const bf16* __restrict__ Vt,
                                                   const int*  __restrict__ mask,
                                                   const int*  __restrict__ anyv,
                                                   bf16* __restrict__ Oh) {
  __shared__ bf16 kst[2][32 * 64];   // K tile: [kv][d], 4 KB per buffer
  __shared__ bf16 vst[2][64 * 32];   // V tile: [d][kv], 4 KB per buffer
  __shared__ bf16 pstage[4][16 * 32];

  const int tid  = threadIdx.x;
  const int wave = tid >> 5;
  const int lane = tid & 31;
  const int l15  = lane & 15;
  const int half = lane >> 4;

  const int bh = blockIdx.y;
  const int b  = bh >> 4;
  const int h  = bh & 15;
  const int q0 = blockIdx.x * 64 + wave * 16;
  const float scale = 0.125f;   // 1/sqrt(64)

  const bf16* Qb = Qh + (size_t)bh * SEQ * HD;
  const bf16* Kb = Kh + (size_t)bh * SEQ * HD;
  const bf16* Vb = Vt + (size_t)bh * HD * SEQ;

  if (!anyv[b]) {   // all-invalid batch: attention output is exactly 0 (uniform exit)
#pragma unroll
    for (int df = 0; df < 4; ++df)
#pragma unroll
      for (int r = 0; r < 8; ++r) {
        const int m = q0 + r + half * 8;
        Oh[(((size_t)(b * SEQ + m)) * NH + h) * HD + df * 16 + l15] = (bf16)0.0f;
      }
    return;
  }

  // stage one kv step: K 32x64 (256 chunks) + V 64x32 (256 chunks), 4/thread
  auto issue = [&](int buf, int kv0) {
#pragma unroll
    for (int c = tid; c < 256; c += 128) {
      const int row = c >> 3, cc = (c & 7) * 8;
      async_copy_b128(&kst[buf][row * 64 + cc], Kb + (size_t)(kv0 + row) * HD + cc);
    }
#pragma unroll
    for (int c = tid; c < 256; c += 128) {
      const int row = c >> 2, cc = (c & 3) * 8;
      async_copy_b128(&vst[buf][row * 32 + cc], Vb + (size_t)row * SEQ + kv0 + cc);
    }
  };

  // Q A-fragments over d (two 32-wide k-steps), resident all loop
  ABf qa[2];
  {
    const bf16* qptr = Qb + (size_t)(q0 + l15) * HD + half * 8;
    qa[0].u[0] = *(const uint4*)(qptr);
    qa[0].u[1] = *(const uint4*)(qptr + 16);
    qa[1].u[0] = *(const uint4*)(qptr + 32);
    qa[1].u[1] = *(const uint4*)(qptr + 48);
  }

  v8f acc[4] = {};
  float mrow[8], lrow[8];
#pragma unroll
  for (int r = 0; r < 8; ++r) { mrow[r] = -__builtin_inff(); lrow[r] = 0.0f; }

  bf16* pw = &pstage[wave][0];
  issue(0, 0);

  for (int kv0 = 0; kv0 < SEQ; kv0 += 32) {
    const int cur = (kv0 >> 5) & 1;
    if (kv0 + 32 < SEQ) { issue(cur ^ 1, kv0 + 32); wait_async_le4(); }
    else                { wait_async_le0(); }
    __syncthreads();

    // ---- S = Q K^T from LDS K tile
    v8f sfr[2] = {};
#pragma unroll
    for (int f = 0; f < 2; ++f) {
      const bf16* kp = &kst[cur][(f * 16 + l15) * 64 + half * 8];
#pragma unroll
      for (int ds = 0; ds < 2; ++ds) {
        ABf bb;
        bb.u[0] = *(const uint4*)(kp + ds * 32);
        bb.u[1] = *(const uint4*)(kp + ds * 32 + 16);
        sfr[f] = wmma_bf16(qa[ds].v, bb.v, sfr[f]);
      }
    }

    float mk[2];
#pragma unroll
    for (int f = 0; f < 2; ++f)
      mk[f] = (mask[b * SEQ + kv0 + f * 16 + l15] != 0) ? 0.0f : -__builtin_inff();

    // ---- online softmax (rows live in 16-lane halves)
#pragma unroll
    for (int r = 0; r < 8; ++r) {
      float s0 = sfr[0][r] * scale + mk[0];
      float s1 = sfr[1][r] * scale + mk[1];
      float tmax = fmaxf(s0, s1);
#pragma unroll
      for (int off = 1; off < 16; off <<= 1) tmax = fmaxf(tmax, __shfl_xor(tmax, off, 32));
      const float mnew = fmaxf(mrow[r], tmax);
      const float corr = (mrow[r] > -__builtin_inff()) ? __expf(mrow[r] - mnew) : 0.0f;
      const float p0 = (s0 > -__builtin_inff()) ? __expf(s0 - mnew) : 0.0f;
      const float p1 = (s1 > -__builtin_inff()) ? __expf(s1 - mnew) : 0.0f;
      float psum = p0 + p1;
#pragma unroll
      for (int off = 1; off < 16; off <<= 1) psum += __shfl_xor(psum, off, 32);
      lrow[r] = lrow[r] * corr + psum;
      mrow[r] = mnew;
#pragma unroll
      for (int df = 0; df < 4; ++df) acc[df][r] = acc[df][r] * corr;
      pw[(r + 8 * half) * 32 + l15]      = (bf16)p0;
      pw[(r + 8 * half) * 32 + 16 + l15] = (bf16)p1;
    }

    wait_ds0();   // wave-private LDS RAW fence for pstage

    // ---- reload P as 16x32 A-fragment
    ABf pa;
    pa.u[0] = *(const uint4*)(pw + l15 * 32 + half * 8);
    pa.u[1] = *(const uint4*)(pw + l15 * 32 + half * 8 + 16);

    // ---- O += P V from LDS V tile
#pragma unroll
    for (int df = 0; df < 4; ++df) {
      const bf16* vp = &vst[cur][(df * 16 + l15) * 32 + half * 8];
      ABf vb;
      vb.u[0] = *(const uint4*)(vp);
      vb.u[1] = *(const uint4*)(vp + 16);
      acc[df] = wmma_bf16(pa.v, vb.v, acc[df]);
    }
    __syncthreads();   // all waves done with this buffer before overwrite
  }

  float inv[8];
#pragma unroll
  for (int r = 0; r < 8; ++r) inv[r] = 1.0f / lrow[r];
#pragma unroll
  for (int df = 0; df < 4; ++df)
#pragma unroll
    for (int r = 0; r < 8; ++r) {
      const int m = q0 + r + half * 8;
      Oh[(((size_t)(b * SEQ + m)) * NH + h) * HD + df * 16 + l15] =
          (bf16)(acc[df][r] * inv[r]);
    }
}

// ---------------------------------------------------------------------------
// host launcher
// ---------------------------------------------------------------------------
extern "C" void kernel_launch(void* const* d_in, const int* in_sizes, int n_in,
                              void* d_out, int out_size, void* d_ws, size_t ws_size,
                              hipStream_t stream) {
  const float* q  = (const float*)d_in[0];
  const float* k  = (const float*)d_in[1];
  const float* v  = (const float*)d_in[2];
  const int*   mk = (const int*)  d_in[3];
  const float* Wq = (const float*)d_in[4];
  const float* bq = (const float*)d_in[5];
  const float* Wk = (const float*)d_in[6];
  const float* bk = (const float*)d_in[7];
  const float* Wv = (const float*)d_in[8];
  const float* bv = (const float*)d_in[9];
  const float* Wo = (const float*)d_in[10];
  const float* bo = (const float*)d_in[11];

  char* ws = (char*)d_ws;
  size_t off = 0;
  auto take = [&](size_t bytes) { char* p = ws + off; off += (bytes + 255) & ~(size_t)255; return p; };

  const size_t ACT   = (size_t)MTOT * EMB;
  const size_t WSZ   = (size_t)EMB * EMB;
  const size_t HEADS = (size_t)BATCH * NH * SEQ * HD;

  bf16* qb  = (bf16*)take(ACT * 2);
  bf16* kb  = (bf16*)take(ACT * 2);
  bf16* vb  = (bf16*)take(ACT * 2);
  bf16* Wqb = (bf16*)take(WSZ * 2);
  bf16* Wkb = (bf16*)take(WSZ * 2);
  bf16* Wvb = (bf16*)take(WSZ * 2);
  bf16* Wob = (bf16*)take(WSZ * 2);
  bf16* Qh  = (bf16*)take(HEADS * 2);
  bf16* Kh  = (bf16*)take(HEADS * 2);
  bf16* Vt  = (bf16*)take(HEADS * 2);
  bf16* Oh  = (bf16*)take(ACT * 2);
  int*  anyv = (int*)take(64);

  cvt_bf16_kernel<<<(ACT / 4 + 255) / 256, 256, 0, stream>>>(q, qb, (int)(ACT / 4));
  cvt_bf16_kernel<<<(ACT / 4 + 255) / 256, 256, 0, stream>>>(k, kb, (int)(ACT / 4));
  cvt_bf16_kernel<<<(ACT / 4 + 255) / 256, 256, 0, stream>>>(v, vb, (int)(ACT / 4));
  cvt_bf16_kernel<<<(WSZ / 4 + 255) / 256, 256, 0, stream>>>(Wq, Wqb, (int)(WSZ / 4));
  cvt_bf16_kernel<<<(WSZ / 4 + 255) / 256, 256, 0, stream>>>(Wk, Wkb, (int)(WSZ / 4));
  cvt_bf16_kernel<<<(WSZ / 4 + 255) / 256, 256, 0, stream>>>(Wv, Wvb, (int)(WSZ / 4));
  cvt_bf16_kernel<<<(WSZ / 4 + 255) / 256, 256, 0, stream>>>(Wo, Wob, (int)(WSZ / 4));

  mask_any_kernel<<<BATCH, 256, 0, stream>>>(mk, anyv);

  dim3 pg(MTOT / 128, EMB / 128);
  proj_gemm_kernel<<<pg, 256, 0, stream>>>(qb, Wqb, bq, Qh, 0);
  proj_gemm_kernel<<<pg, 256, 0, stream>>>(kb, Wkb, bk, Kh, 0);
  proj_gemm_kernel<<<pg, 256, 0, stream>>>(vb, Wvb, bv, Vt, 1);

  attn_kernel<<<dim3(SEQ / 64, BATCH * NH), 128, 0, stream>>>(Qh, Kh, Vt, mk, anyv, Oh);

  proj_gemm_kernel<<<pg, 256, 0, stream>>>(Oh, Wob, bo, d_out, 2);
}
